// GroupMixer_88364657148003
// MI455X (gfx1250) — matrix operand order repaired and checked by reading, compiled, not run on hardware
//
#include <hip/hip_runtime.h>
#include <hip/hip_bf16.h>
#include <cstdint>

// ---------------- problem constants ----------------
constexpr int B_  = 64, T_ = 128, A_ = 30, G_ = 3, E_ = 64, H_ = 256;
constexpr int FA_ = 32, FE_ = 16, NE_ = 30;
constexpr int SD_ = A_ * FA_ + NE_ * FE_ + A_ * 36;   // 2520
constexpr int N_  = B_ * T_;                          // 8192
constexpr int SDP = 2528;                             // SD padded to mult of 32

typedef unsigned short u16;
typedef __attribute__((ext_vector_type(16))) __bf16 v16bf;
typedef __attribute__((ext_vector_type(8)))  float  v8f;
typedef __attribute__((ext_vector_type(8)))  u16    v8us;

__device__ __forceinline__ u16 f2bf(float f) {
  unsigned u = __builtin_bit_cast(unsigned, f);
  unsigned r = u + 0x7fffu + ((u >> 16) & 1u);          // RNE
  return (u16)(r >> 16);
}
__device__ __forceinline__ float bf2f(u16 h) {
  unsigned u = ((unsigned)h) << 16;
  return __builtin_bit_cast(float, u);
}

// CDNA5 direct HBM -> LDS async copy (16B per lane), tracked by ASYNCcnt.
__device__ __forceinline__ void async_ld_b128(unsigned lds_off, const void* gptr) {
  asm volatile("global_load_async_to_lds_b128 %0, %1, off"
               :: "v"(lds_off), "v"((unsigned long long)(uintptr_t)gptr)
               : "memory");
}
__device__ __forceinline__ void wait_async0() {
  asm volatile("s_wait_asynccnt 0" ::: "memory");
}

// ---------------- conversion kernels ----------------
__global__ void cvt_states(const float* __restrict__ src, u16* __restrict__ dst) {
  size_t i = (size_t)blockIdx.x * blockDim.x + threadIdx.x;
  size_t total = (size_t)N_ * SDP;
  if (i >= total) return;
  int n = (int)(i / SDP), k = (int)(i % SDP);
  dst[i] = (k < SD_) ? f2bf(src[(size_t)n * SD_ + k]) : (u16)0;
}

// W (K, M) f32 row-major -> Wt (Mp, Kp) bf16 row-major (transposed, zero padded)
__global__ void cvt_wT(const float* __restrict__ src, u16* __restrict__ dst,
                       int K, int M, int Kp, int Mp) {
  size_t i = (size_t)blockIdx.x * blockDim.x + threadIdx.x;
  size_t total = (size_t)Mp * Kp;
  if (i >= total) return;
  int m = (int)(i / Kp), k = (int)(i % Kp);
  dst[i] = (m < M && k < K) ? f2bf(src[(size_t)k * M + m]) : (u16)0;
}

// ---------------- WMMA bf16 GEMM ----------------
// C(N x Mp) = act( A(N x K) @ Wt(Mp x K)^T + bias ), bias real length Mreal.
// OUT: 0=f32 store, 1=bf16 store.  ACT: 0 none, 1 relu, 2 abs.
// 128x128 block tile, 4 wave32s each 64x64 (4x4 wmma frags), K step 32,
// double-buffered 32 KB LDS fed by GLOBAL_LOAD_ASYNC_TO_LDS_B128.
template <int OUT, int ACT>
__global__ __launch_bounds__(128)
void gemm_wmma_bf16(const u16* __restrict__ Amat, const u16* __restrict__ Wt,
                    const float* __restrict__ bias, void* __restrict__ outp,
                    int K, int Mp, int Mreal) {
  __shared__ __align__(16) u16 As[2][128 * 32];
  __shared__ __align__(16) u16 Bs[2][128 * 32];

  const int tid  = threadIdx.x;
  const int lane = tid & 31, wid = tid >> 5;
  const int wr = wid & 1;        // row half (64 rows)
  const int wc = wid >> 1;       // col half (64 cols)
  const int lr = lane & 15, hh = lane >> 4;
  const int row0 = blockIdx.y * 128;
  const int col0 = blockIdx.x * 128;

  v8f acc[4][4] = {};

  // Per-thread tile feed: thread t streams row t of the A tile and row t of
  // the B (W^T) tile — 4 async 16B transfers each, direct to LDS.
  const u16* aRow = Amat + (size_t)(row0 + tid) * K;
  const u16* bRow = Wt   + (size_t)(col0 + tid) * K;
  const unsigned ldsA0 = (unsigned)(uintptr_t)&As[0][tid * 32];
  const unsigned ldsA1 = (unsigned)(uintptr_t)&As[1][tid * 32];
  const unsigned ldsB0 = (unsigned)(uintptr_t)&Bs[0][tid * 32];
  const unsigned ldsB1 = (unsigned)(uintptr_t)&Bs[1][tid * 32];

  auto issue_tile = [&](int bufi, int kb) {
    const unsigned la = bufi ? ldsA1 : ldsA0;
    const unsigned lb = bufi ? ldsB1 : ldsB0;
#pragma unroll
    for (int j = 0; j < 4; ++j) {
      async_ld_b128(la + j * 16, aRow + kb + j * 8);
      async_ld_b128(lb + j * 16, bRow + kb + j * 8);
    }
  };

  issue_tile(0, 0);

  for (int kb = 0; kb < K; kb += 32) {
    const int buf = (kb >> 5) & 1;
    wait_async0();        // own async writes for `buf` complete
    __syncthreads();      // everyone's writes complete -> safe to read `buf`
    if (kb + 32 < K) issue_tile(buf ^ 1, kb + 32);

    union FragU { v16bf v; v8us h[2]; };
    FragU af[4], bfr[4];
#pragma unroll
    for (int r = 0; r < 4; ++r) {
      // ISA 16-bit A 16x32: lanes0-15 hold K{0..7,16..23}; lanes16-31 K{8..15,24..31}
      int row = wr * 64 + r * 16 + lr;
      af[r].h[0] = *(const v8us*)&As[buf][row * 32 + hh * 8];
      af[r].h[1] = *(const v8us*)&As[buf][row * 32 + 16 + hh * 8];
    }
#pragma unroll
    for (int c = 0; c < 4; ++c) {
      // ISA 16-bit B 32x16: lanes0-15 (N=lane) K{0..7,16..23}; lanes16-31 K{8..15,24..31}
      int crow = wc * 64 + c * 16 + lr;
      bfr[c].h[0] = *(const v8us*)&Bs[buf][crow * 32 + hh * 8];
      bfr[c].h[1] = *(const v8us*)&Bs[buf][crow * 32 + 16 + hh * 8];
    }
#pragma unroll
    for (int r = 0; r < 4; ++r)
#pragma unroll
      for (int c = 0; c < 4; ++c)
        acc[r][c] = __builtin_amdgcn_wmma_f32_16x16x32_bf16(
            false, af[r].v, false, bfr[c].v, (short)0, acc[r][c], false, false);
  }

#pragma unroll
  for (int r = 0; r < 4; ++r)
#pragma unroll
    for (int c = 0; c < 4; ++c) {
      const int colg = col0 + wc * 64 + c * 16 + lr;
      const float bv = (colg < Mreal) ? bias[colg] : 0.f;
      const int rowb = row0 + wr * 64 + r * 16 + hh * 8;
#pragma unroll
      for (int v = 0; v < 8; ++v) {
        float x = acc[r][c][v] + bv;
        if (ACT == 1) x = x > 0.f ? x : 0.f;
        if (ACT == 2) x = fabsf(x);
        size_t ix = (size_t)(rowb + v) * Mp + colg;
        if (OUT == 0) ((float*)outp)[ix] = x;
        else          ((u16*)outp)[ix]   = f2bf(x);
      }
    }
}

// ---------------- combine stage 1: per (g, n) ----------------
__global__ __launch_bounds__(64)
void combine_group(const float* __restrict__ qvals, const float* __restrict__ states,
                   const u16* __restrict__ w1, const float* __restrict__ b1,
                   const float* __restrict__ w2, const float* __restrict__ hgb2,
                   const float* __restrict__ gb2W2, const float* __restrict__ gb2b2,
                   float* __restrict__ gq) {
  const int n = blockIdx.x, g = blockIdx.y, e = threadIdx.x;
  const int b = n / T_, t = n - b * T_;
  __shared__ float mq[32];
  __shared__ float red[64];
  if (e < A_) {
    float v = states[(size_t)b * T_ * SD_ + e * FA_ + (FA_ - G_) + g];
    float mask = (v == 1.0f) ? 1.f : 0.f;
    mq[e] = qvals[((size_t)b * T_ + t) * A_ + e] * mask;
  }
  __syncthreads();
  const u16* w1p = w1 + ((size_t)g * N_ + n) * 1920;
  float acc = b1[((size_t)g * N_ + n) * 128 + e];
#pragma unroll 6
  for (int a = 0; a < A_; ++a) acc = fmaf(mq[a], bf2f(w1p[a * 64 + e]), acc);
  float hid = acc > 0.f ? acc : (__expf(acc) - 1.f);   // elu
  size_t ix = ((size_t)g * N_ + n) * 128 + e;
  red[e] = hid * w2[ix] + hgb2[ix] * gb2W2[g * 64 + e];
  __syncthreads();
  for (int s = 32; s > 0; s >>= 1) {
    if (e < s) red[e] += red[e + s];
    __syncthreads();
  }
  if (e == 0) gq[(size_t)n * G_ + g] = red[0] + gb2b2[g];
}

// ---------------- combine stage 2: per n ----------------
__global__ __launch_bounds__(64)
void combine_final(const float* __restrict__ gq, const float* __restrict__ hw1,
                   const float* __restrict__ hb1, const float* __restrict__ hw2,
                   const float* __restrict__ hhb2, const float* __restrict__ hb2W2,
                   const float* __restrict__ hb2b2, float* __restrict__ out) {
  const int n = blockIdx.x, e = threadIdx.x;
  __shared__ float red[64];
  float acc = hb1[(size_t)n * 128 + e];
#pragma unroll
  for (int g = 0; g < G_; ++g)
    acc = fmaf(gq[(size_t)n * G_ + g], hw1[(size_t)n * 256 + g * 64 + e], acc);
  float hid = acc > 0.f ? acc : (__expf(acc) - 1.f);
  red[e] = hid * hw2[(size_t)n * 128 + e] + hhb2[(size_t)n * 128 + e] * hb2W2[e];
  __syncthreads();
  for (int s = 32; s > 0; s >>= 1) {
    if (e < s) red[e] += red[e + s];
    __syncthreads();
  }
  if (e == 0) out[n] = red[0] + hb2b2[0];
}

// ---------------- host launcher ----------------
extern "C" void kernel_launch(void* const* d_in, const int* in_sizes, int n_in,
                              void* d_out, int out_size, void* d_ws, size_t ws_size,
                              hipStream_t stream) {
  (void)in_sizes; (void)n_in; (void)out_size; (void)ws_size;
  const float* qvals  = (const float*)d_in[0];
  const float* states = (const float*)d_in[1];

  char* ws = (char*)d_ws;
  size_t off = 0;
  auto alloc = [&](size_t bytes) -> void* {
    void* p = ws + off;
    off += (bytes + 255) & ~(size_t)255;
    return p;
  };

  // bf16 staging (per-group logical tensors contiguous across g).
  u16* sbf = (u16*)alloc((size_t)N_ * SDP * 2);
  u16* wt_gw1 = (u16*)alloc((size_t)3 * 256 * SDP * 2);   // [g][256][SDP]
  u16* wt_gb1 = (u16*)alloc((size_t)3 * 256 * SDP * 2);
  u16* wt_gw2 = (u16*)alloc((size_t)3 * 256 * SDP * 2);
  u16* wt_gb2 = (u16*)alloc((size_t)3 * 128 * SDP * 2);   // [g][128][SDP]
  u16* wt_hw1 = (u16*)alloc((size_t)256 * SDP * 2);
  u16* wt_hb1 = (u16*)alloc((size_t)256 * SDP * 2);
  u16* wt_hw2 = (u16*)alloc((size_t)256 * SDP * 2);
  u16* wt_hb2 = (u16*)alloc((size_t)128 * SDP * 2);
  u16* wt2_gw1 = (u16*)alloc((size_t)3 * 1920 * 256 * 2);
  u16* wt2_gb1 = (u16*)alloc((size_t)3 * 128 * 256 * 2);
  u16* wt2_gw2 = (u16*)alloc((size_t)3 * 128 * 256 * 2);
  u16* wt2_hw1 = (u16*)alloc((size_t)256 * 256 * 2);
  u16* wt2_hb1 = (u16*)alloc((size_t)128 * 256 * 2);
  u16* wt2_hw2 = (u16*)alloc((size_t)128 * 256 * 2);
  u16* hid_gw1 = (u16*)alloc((size_t)3 * N_ * 256 * 2);   // [g][N][256]
  u16* hid_gb1 = (u16*)alloc((size_t)3 * N_ * 256 * 2);
  u16* hid_gw2 = (u16*)alloc((size_t)3 * N_ * 256 * 2);
  u16* hid_hw1 = (u16*)alloc((size_t)N_ * 256 * 2);
  u16* hid_hb1 = (u16*)alloc((size_t)N_ * 256 * 2);
  u16* hid_hw2 = (u16*)alloc((size_t)N_ * 256 * 2);
  float* hgb2 = (float*)alloc((size_t)3 * N_ * 128 * 4); // [g][N][128]
  u16*   w1o  = (u16*)alloc((size_t)3 * N_ * 1920 * 2);  // [g][N][1920] bf16
  float* b1o  = (float*)alloc((size_t)3 * N_ * 128 * 4);
  float* w2o  = (float*)alloc((size_t)3 * N_ * 128 * 4);
  float* hhb2 = (float*)alloc((size_t)N_ * 128 * 4);
  float* hw1o = (float*)alloc((size_t)N_ * 256 * 4);
  float* hb1o = (float*)alloc((size_t)N_ * 128 * 4);
  float* hw2o = (float*)alloc((size_t)N_ * 128 * 4);
  float* gqv  = (float*)alloc((size_t)N_ * G_ * 4);

  // ---- conversions ----
  cvt_states<<<(unsigned)(((size_t)N_ * SDP) / 256), 256, 0, stream>>>(states, sbf);
  auto cvtW = [&](const float* src, u16* dst, int K, int M, int Kp, int Mp) {
    cvt_wT<<<(unsigned)(((size_t)Mp * Kp) / 256), 256, 0, stream>>>(src, dst, K, M, Kp, Mp);
  };
  for (int g = 0; g < 3; ++g) {
    cvtW((const float*)d_in[2]  + (size_t)g * SD_ * H_, wt_gw1 + (size_t)g * 256 * SDP, SD_, 256, SDP, 256);
    cvtW((const float*)d_in[6]  + (size_t)g * SD_ * H_, wt_gb1 + (size_t)g * 256 * SDP, SD_, 256, SDP, 256);
    cvtW((const float*)d_in[10] + (size_t)g * SD_ * H_, wt_gw2 + (size_t)g * 256 * SDP, SD_, 256, SDP, 256);
    cvtW((const float*)d_in[14] + (size_t)g * SD_ * E_, wt_gb2 + (size_t)g * 128 * SDP, SD_, 64, SDP, 128);
    cvtW((const float*)d_in[4]  + (size_t)g * H_ * 1920, wt2_gw1 + (size_t)g * 1920 * 256, 256, 1920, 256, 1920);
    cvtW((const float*)d_in[8]  + (size_t)g * H_ * E_,   wt2_gb1 + (size_t)g * 128 * 256, 256, 64, 256, 128);
    cvtW((const float*)d_in[12] + (size_t)g * H_ * E_,   wt2_gw2 + (size_t)g * 128 * 256, 256, 64, 256, 128);
  }
  cvtW((const float*)d_in[18], wt_hw1, SD_, 256, SDP, 256);
  cvtW((const float*)d_in[22], wt_hb1, SD_, 256, SDP, 256);
  cvtW((const float*)d_in[26], wt_hw2, SD_, 256, SDP, 256);
  cvtW((const float*)d_in[30], wt_hb2, SD_, 64, SDP, 128);
  cvtW((const float*)d_in[20], wt2_hw1, 256, 192, 256, 256);
  cvtW((const float*)d_in[24], wt2_hb1, 256, 64, 256, 128);
  cvtW((const float*)d_in[28], wt2_hw2, 256, 64, 256, 128);

  // ---- first-layer GEMMs: relu(s @ W1 + b1) ----
  const dim3 blk(128);
  auto grid_of = [](int Mp) { return dim3((unsigned)(Mp / 128), N_ / 128); };
  for (int g = 0; g < 3; ++g) {
    gemm_wmma_bf16<1, 1><<<grid_of(256), blk, 0, stream>>>(
        sbf, wt_gw1 + (size_t)g * 256 * SDP, (const float*)d_in[3] + g * H_,
        hid_gw1 + (size_t)g * N_ * 256, SDP, 256, 256);
    gemm_wmma_bf16<1, 1><<<grid_of(256), blk, 0, stream>>>(
        sbf, wt_gb1 + (size_t)g * 256 * SDP, (const float*)d_in[7] + g * H_,
        hid_gb1 + (size_t)g * N_ * 256, SDP, 256, 256);
    gemm_wmma_bf16<1, 1><<<grid_of(256), blk, 0, stream>>>(
        sbf, wt_gw2 + (size_t)g * 256 * SDP, (const float*)d_in[11] + g * H_,
        hid_gw2 + (size_t)g * N_ * 256, SDP, 256, 256);
    gemm_wmma_bf16<0, 1><<<grid_of(128), blk, 0, stream>>>(
        sbf, wt_gb2 + (size_t)g * 128 * SDP, (const float*)d_in[15] + g * E_,
        hgb2 + (size_t)g * N_ * 128, SDP, 128, 64);
  }
  gemm_wmma_bf16<1, 1><<<grid_of(256), blk, 0, stream>>>(
      sbf, wt_hw1, (const float*)d_in[19], hid_hw1, SDP, 256, 256);
  gemm_wmma_bf16<1, 1><<<grid_of(256), blk, 0, stream>>>(
      sbf, wt_hb1, (const float*)d_in[23], hid_hb1, SDP, 256, 256);
  gemm_wmma_bf16<1, 1><<<grid_of(256), blk, 0, stream>>>(
      sbf, wt_hw2, (const float*)d_in[27], hid_hw2, SDP, 256, 256);
  gemm_wmma_bf16<0, 1><<<grid_of(128), blk, 0, stream>>>(
      sbf, wt_hb2, (const float*)d_in[31], hhb2, SDP, 128, 64);

  // ---- second-layer GEMMs ----
  for (int g = 0; g < 3; ++g) {
    gemm_wmma_bf16<1, 2><<<grid_of(1920), blk, 0, stream>>>(
        hid_gw1 + (size_t)g * N_ * 256, wt2_gw1 + (size_t)g * 1920 * 256,
        (const float*)d_in[5] + g * 1920, w1o + (size_t)g * N_ * 1920, 256, 1920, 1920);
    gemm_wmma_bf16<0, 0><<<grid_of(128), blk, 0, stream>>>(
        hid_gb1 + (size_t)g * N_ * 256, wt2_gb1 + (size_t)g * 128 * 256,
        (const float*)d_in[9] + g * E_, b1o + (size_t)g * N_ * 128, 256, 128, 64);
    gemm_wmma_bf16<0, 2><<<grid_of(128), blk, 0, stream>>>(
        hid_gw2 + (size_t)g * N_ * 256, wt2_gw2 + (size_t)g * 128 * 256,
        (const float*)d_in[13] + g * E_, w2o + (size_t)g * N_ * 128, 256, 128, 64);
  }
  gemm_wmma_bf16<0, 2><<<grid_of(256), blk, 0, stream>>>(
      hid_hw1, wt2_hw1, (const float*)d_in[21], hw1o, 256, 256, 192);
  gemm_wmma_bf16<0, 0><<<grid_of(128), blk, 0, stream>>>(
      hid_hb1, wt2_hb1, (const float*)d_in[25], hb1o, 256, 128, 64);
  gemm_wmma_bf16<0, 2><<<grid_of(128), blk, 0, stream>>>(
      hid_hw2, wt2_hw2, (const float*)d_in[29], hw2o, 256, 128, 64);

  // ---- combine ----
  combine_group<<<dim3(N_, G_), dim3(64), 0, stream>>>(
      qvals, states, w1o, b1o, w2o, hgb2,
      (const float*)d_in[16], (const float*)d_in[17], gqv);
  combine_final<<<dim3(N_), dim3(64), 0, stream>>>(
      gqv, hw1o, hb1o, hw2o, hhb2,
      (const float*)d_in[32], (const float*)d_in[33], (float*)d_out);
}
